// MultiHeadAttention_37082747634030
// MI455X (gfx1250) — compile-verified, matrix operand order checked by reading, not run on hardware
//
#include <hip/hip_runtime.h>

// ---------------------------------------------------------------------------
// Multi-head attention forward for gfx1250 (CDNA5), bf16 WMMA pipeline.
//   B=8, N=1024, E=768, H=8, D=96
// ---------------------------------------------------------------------------

typedef __attribute__((ext_vector_type(16))) __bf16 v16bf;
typedef __attribute__((ext_vector_type(8)))  float  v8f;

#define Ed   768
#define QKVd 2304
#define Bd   8
#define Nd   1024
#define Hd   8
#define Dd   96
#define BN   8192    // B*N

static __device__ __forceinline__ v8f wmma_bf16(v16bf a, v16bf b, v8f c) {
  // D = A(16x32 bf16) * B(32x16 bf16) + C(16x16 f32)
  return __builtin_amdgcn_wmma_f32_16x16x32_bf16(
      /*neg_a=*/false, a, /*neg_b=*/false, b,
      /*c_mod=*/(short)0, c, /*reuse_a=*/false, /*reuse_b=*/false);
}

// ---------------------------------------------------------------------------
// fp32 -> bf16 conversions (weights transposed so B-fragment column reads are
// contiguous 16-element runs per lane)
// ---------------------------------------------------------------------------
__global__ __launch_bounds__(256) void cvt_kernel(const float* __restrict__ in,
                                                  __bf16* __restrict__ out, int n) {
  int i = blockIdx.x * 256 + threadIdx.x;
  if (i < n) out[i] = (__bf16)in[i];
}

__global__ __launch_bounds__(256) void cvt_transpose_kernel(const float* __restrict__ in,
                                                            __bf16* __restrict__ out,
                                                            int rows, int cols) {
  // in: rows x cols row-major; out: cols x rows (out[c][r] = in[r][c])
  int i = blockIdx.x * 256 + threadIdx.x;
  if (i < rows * cols) {
    int r = i / cols, c = i % cols;
    out[(size_t)c * rows + r] = (__bf16)in[i];
  }
}

// ---------------------------------------------------------------------------
// QKV GEMM: C(8192 x 2304) = xb @ Wqkv + b_qkv, de-interleaved into
//   Q(B,H,N,D), K(B,H,N,D), V^T(B,H,D,N)   (qkv index is innermost in cols)
// One wave computes a 32x64 tile (2 M-subtiles x 4 N-subtiles; 8 WMMA and
// 6 fragment loads per K-step -> 0.75 frag-loads/WMMA).
// ---------------------------------------------------------------------------
__global__ __launch_bounds__(32) void qkv_gemm_kernel(
    const __bf16* __restrict__ xb, const __bf16* __restrict__ WqkvT,
    const float* __restrict__ bias,
    __bf16* __restrict__ Qb, __bf16* __restrict__ Kb, __bf16* __restrict__ VTb) {
  const int lane = threadIdx.x;
  const int row  = lane & 15;
  const int half = lane >> 4;
  const int m0 = blockIdx.y * 32;
  const int n0 = blockIdx.x * 64;

  v8f acc[2][4] = {};
  const __bf16* ap0 = xb    + (size_t)(m0 + row) * Ed + half * 16;
  const __bf16* ap1 = ap0 + (size_t)16 * Ed;
  const __bf16* bp  = WqkvT + (size_t)(n0 + row) * Ed + half * 16;
  for (int kb = 0; kb < Ed; kb += 32) {
    v16bf a0 = *(const v16bf*)(ap0 + kb);
    v16bf a1 = *(const v16bf*)(ap1 + kb);
#pragma unroll
    for (int t = 0; t < 4; ++t) {
      v16bf b = *(const v16bf*)(bp + (size_t)t * 16 * Ed + kb);
      acc[0][t] = wmma_bf16(a0, b, acc[0][t]);
      acc[1][t] = wmma_bf16(a1, b, acc[1][t]);
    }
  }

#pragma unroll
  for (int t = 0; t < 4; ++t) {
    const int cg = n0 + t * 16 + row;
    const float bv = bias[cg];
    const int s  = cg % 3;      // 0=Q 1=K 2=V (qkv index innermost)
    const int hd = cg / 3;
    const int h  = hd / Dd;
    const int dd = hd % Dd;
#pragma unroll
    for (int mt = 0; mt < 2; ++mt) {
      const int mbase = m0 + mt * 16 + half * 8;  // C frag: row = 8*half + j
#pragma unroll
      for (int j = 0; j < 8; ++j) {
        const int m  = mbase + j;
        const int bb = m >> 10;
        const int nn = m & (Nd - 1);
        const __bf16 val = (__bf16)(acc[mt][t][j] + bv);
        const size_t bh = (size_t)bb * Hd + h;
        if (s == 0)      Qb[(bh * Nd + nn) * Dd + dd]  = val;
        else if (s == 1) Kb[(bh * Nd + nn) * Dd + dd]  = val;
        else             VTb[(bh * Dd + dd) * Nd + nn] = val;
      }
    }
  }
}

// ---------------------------------------------------------------------------
// Attention: one workgroup (8 waves) per 16 query rows of one (b,h).
//   Phase 1: S(16x1024) = Q @ K^T      (waves tile keys, 3 WMMA per 16x16)
//   Phase 2: row softmax, fold 1/sqrt(E), store P bf16 in-place in LDS
//   Phase 3: O(16x96) = P @ V          (6 waves, 32 WMMA each, K=1024)
// ---------------------------------------------------------------------------
__global__ __launch_bounds__(256) void attn_kernel(
    const __bf16* __restrict__ Qb, const __bf16* __restrict__ Kb,
    const __bf16* __restrict__ VTb, __bf16* __restrict__ attnb) {
  __shared__ alignas(64) __bf16 S[16 * Nd];   // 32 KB: energies -> probabilities
  __shared__ float red[256];

  const int tid  = threadIdx.x;
  const int lane = tid & 31;
  const int wave = tid >> 5;
  const int row  = lane & 15;
  const int half = lane >> 4;

  const int q0 = blockIdx.x * 16;
  const int bh = blockIdx.y;       // b*H + h
  const int b  = bh >> 3;
  const int h  = bh & 7;

  // ---- Phase 1: energies for key columns [wave*128, wave*128+128) ----
  const __bf16* qp = Qb + ((size_t)bh * Nd + q0 + row) * Dd + half * 16;
  v16bf aq0 = *(const v16bf*)(qp);
  v16bf aq1 = *(const v16bf*)(qp + 32);
  v16bf aq2 = *(const v16bf*)(qp + 64);
#pragma unroll
  for (int ct = 0; ct < 8; ++ct) {
    const int kc = wave * 128 + ct * 16;
    const __bf16* kp = Kb + ((size_t)bh * Nd + kc + row) * Dd + half * 16;
    v16bf bk0 = *(const v16bf*)(kp);
    v16bf bk1 = *(const v16bf*)(kp + 32);
    v16bf bk2 = *(const v16bf*)(kp + 64);
    v8f acc = {};
    acc = wmma_bf16(aq0, bk0, acc);
    acc = wmma_bf16(aq1, bk1, acc);
    acc = wmma_bf16(aq2, bk2, acc);
    const int scol = kc + row;
#pragma unroll
    for (int j = 0; j < 8; ++j)
      S[(half * 8 + j) * Nd + scol] = (__bf16)acc[j];
  }
  __syncthreads();

  // ---- Phase 2: softmax over each row (1024 entries), 16 threads per row ----
  const int r = tid >> 4;
  const int c = tid & 15;
  float pm = -3.0e38f;
  for (int i = 0; i < 64; ++i)
    pm = fmaxf(pm, (float)S[r * Nd + c + 16 * i]);
  red[tid] = pm;
  __syncthreads();
  float rmax = -3.0e38f;
#pragma unroll
  for (int j = 0; j < 16; ++j) rmax = fmaxf(rmax, red[r * 16 + j]);
  __syncthreads();                       // red reads done before reuse

  float ps = 0.f;
  for (int i = 0; i < 64; ++i) {
    const int idx = r * Nd + c + 16 * i;
    const float v = __expf((float)S[idx] - rmax);
    S[idx] = (__bf16)v;                  // own elements only
    ps += v;
  }
  red[tid] = ps;
  __syncthreads();
  float rsum = 0.f;
#pragma unroll
  for (int j = 0; j < 16; ++j) rsum += red[r * 16 + j];
  const float scale = 1.0f / (rsum * 27.712812921102035f);  // softmax / sqrt(768)
  for (int i = 0; i < 64; ++i) {
    const int idx = r * Nd + c + 16 * i;
    S[idx] = (__bf16)((float)S[idx] * scale);
  }
  __syncthreads();

  // ---- Phase 3: O = P @ V ; 6 waves each own one 16-wide slice of D=96 ----
  if (wave < 6) {
    const int dd0 = wave * 16;
    v8f acc = {};
    const __bf16* vp = VTb + ((size_t)bh * Dd + dd0 + row) * Nd + half * 16;
    for (int kc = 0; kc < Nd; kc += 32) {
      v16bf a  = *(const v16bf*)&S[row * Nd + kc + half * 16];
      v16bf bv = *(const v16bf*)(vp + kc);
      acc = wmma_bf16(a, bv, acc);
    }
    const int col = h * Dd + dd0 + row;           // merged-head E index
#pragma unroll
    for (int j = 0; j < 8; ++j) {
      const int m = q0 + half * 8 + j;            // query row within N
      attnb[((size_t)(b * Nd + m)) * Ed + col] = (__bf16)acc[j];
    }
  }
}

// ---------------------------------------------------------------------------
// Output projection: out(8192 x 768) f32 = attn_bf16 @ Wproj + b_proj
// One wave computes a 32x64 tile.
// ---------------------------------------------------------------------------
__global__ __launch_bounds__(32) void proj_gemm_kernel(
    const __bf16* __restrict__ ab, const __bf16* __restrict__ WprojT,
    const float* __restrict__ bias, float* __restrict__ out) {
  const int lane = threadIdx.x;
  const int row  = lane & 15;
  const int half = lane >> 4;
  const int m0 = blockIdx.y * 32;
  const int n0 = blockIdx.x * 64;

  v8f acc[2][4] = {};
  const __bf16* ap0 = ab     + (size_t)(m0 + row) * Ed + half * 16;
  const __bf16* ap1 = ap0 + (size_t)16 * Ed;
  const __bf16* bp  = WprojT + (size_t)(n0 + row) * Ed + half * 16;
  for (int kb = 0; kb < Ed; kb += 32) {
    v16bf a0 = *(const v16bf*)(ap0 + kb);
    v16bf a1 = *(const v16bf*)(ap1 + kb);
#pragma unroll
    for (int t = 0; t < 4; ++t) {
      v16bf b = *(const v16bf*)(bp + (size_t)t * 16 * Ed + kb);
      acc[0][t] = wmma_bf16(a0, b, acc[0][t]);
      acc[1][t] = wmma_bf16(a1, b, acc[1][t]);
    }
  }
#pragma unroll
  for (int t = 0; t < 4; ++t) {
    const int cg = n0 + t * 16 + row;
    const float bv = bias[cg];
#pragma unroll
    for (int mt = 0; mt < 2; ++mt) {
      const int mbase = m0 + mt * 16 + half * 8;
#pragma unroll
      for (int j = 0; j < 8; ++j)
        out[(size_t)(mbase + j) * Ed + cg] = acc[mt][t][j] + bv;
    }
  }
}

// ---------------------------------------------------------------------------
extern "C" void kernel_launch(void* const* d_in, const int* in_sizes, int n_in,
                              void* d_out, int out_size, void* d_ws, size_t ws_size,
                              hipStream_t stream) {
  const float* x      = (const float*)d_in[0];   // (B,N,E)
  const float* W_qkv  = (const float*)d_in[1];   // (E, 3E)
  const float* b_qkv  = (const float*)d_in[2];   // (3E,)
  const float* W_proj = (const float*)d_in[3];   // (E, E)
  const float* b_proj = (const float*)d_in[4];   // (E,)
  float* out = (float*)d_out;

  char* ws = (char*)d_ws;
  size_t off = 0;
  auto take = [&](size_t bytes) -> void* {
    off = (off + 255) & ~(size_t)255;
    void* p = ws + off;
    off += bytes;
    return p;
  };
  __bf16* xb     = (__bf16*)take((size_t)BN * Ed * 2);
  __bf16* WqkvT  = (__bf16*)take((size_t)QKVd * Ed * 2);
  __bf16* WprojT = (__bf16*)take((size_t)Ed * Ed * 2);
  __bf16* Qb     = (__bf16*)take((size_t)Bd * Hd * Nd * Dd * 2);
  __bf16* Kb     = (__bf16*)take((size_t)Bd * Hd * Nd * Dd * 2);
  __bf16* VTb    = (__bf16*)take((size_t)Bd * Hd * Dd * Nd * 2);
  __bf16* attnb  = (__bf16*)take((size_t)BN * Ed * 2);

  // 1) conversions
  {
    int n = BN * Ed;
    cvt_kernel<<<(n + 255) / 256, 256, 0, stream>>>(x, xb, n);
  }
  cvt_transpose_kernel<<<(Ed * QKVd + 255) / 256, 256, 0, stream>>>(W_qkv, WqkvT, Ed, QKVd);
  cvt_transpose_kernel<<<(Ed * Ed + 255) / 256, 256, 0, stream>>>(W_proj, WprojT, Ed, Ed);

  // 2) QKV projection + de-interleave into Q, K, V^T
  qkv_gemm_kernel<<<dim3(QKVd / 64, BN / 32), 32, 0, stream>>>(
      xb, WqkvT, b_qkv, Qb, Kb, VTb);

  // 3) attention (64 q-tiles x 64 (b,h) pairs)
  attn_kernel<<<dim3(Nd / 16, Bd * Hd), 256, 0, stream>>>(Qb, Kb, VTb, attnb);

  // 4) output projection
  proj_gemm_kernel<<<dim3(Ed / 64, BN / 32), 32, 0, stream>>>(
      attnb, WprojT, b_proj, out);
}
